// EnhancedAttentionLayer_15582141350417
// MI455X (gfx1250) — compile-verified, matrix operand order checked by reading, not run on hardware
//
#include <hip/hip_runtime.h>

// Problem constants (B, S, D from the reference)
#define B_ 4
#define S_ 2048
#define D_ 1024

typedef __attribute__((ext_vector_type(16))) __bf16        v16bf;
typedef __attribute__((ext_vector_type(8)))  float         v8f;
typedef __attribute__((ext_vector_type(4)))  unsigned int  v4u;
typedef __attribute__((ext_vector_type(4)))  float         v4f;

// Explicit global (addrspace(1)) pointee types so loads/stores lower to
// global_load_b128 / global_store_* instead of flat_* (flat ties up the LDS
// issue path and counts against both LOADcnt and DScnt on CDNA5).
typedef __attribute__((address_space(1))) const unsigned short gc_us;
typedef __attribute__((address_space(1)))       unsigned short g_us;
typedef __attribute__((address_space(1))) const v4u            gc_v4u;
typedef __attribute__((address_space(1)))       v4u            g_v4u;
typedef __attribute__((address_space(1))) const v4f            gc_v4f;
typedef __attribute__((address_space(1))) const float          gc_f32;
typedef __attribute__((address_space(1)))       float          g_f32;

__device__ __forceinline__ unsigned short f2bf(float f) {
    unsigned int u = __float_as_uint(f);
    u += 0x7FFFu + ((u >> 16) & 1u);      // round-to-nearest-even
    return (unsigned short)(u >> 16);
}
__device__ __forceinline__ float bf2f(unsigned short h) {
    return __uint_as_float(((unsigned int)h) << 16);
}

union Frag { v4u q[2]; v16bf v; };

// Load one 16x32 bf16 WMMA fragment given a per-lane global base pointer that
// already encodes row and the lane's K-phase (lane<16 -> +0, lane>=16 -> +8).
// elemOff is a compile-time constant so both 16B loads fold into immediate
// offsets of the same address register.
__device__ __forceinline__ v16bf load_frag_g(gc_us* __restrict__ p, int elemOff) {
    gc_v4u* q = (gc_v4u*)(p + elemOff);
    Frag f;
    f.q[0] = q[0];   // K = kb .. kb+7
    f.q[1] = q[2];   // K = kb+16 .. kb+23  (+16 bf16 elements)
    return f.v;
}

// -------------------------------------------------------------------------
// Generic bf16 WMMA GEMM:  C[m,n] = scale * sum_k A[m,k]*Bm[n,k]  (+ bias[n])
// Block = 256 threads = 8 waves (2 x 4); block tile 128x128; wave tile 64x32.
// Inner loop is double-buffered over K (two fragment register sets, unroll 2)
// so global_load_b128 for step t+1 overlap the 8 WMMAs of step t.
// OUT_BF16: store bf16, else fp32.  TRANS: store C transposed per batch
// (used to produce V^T directly).  BIAS: add bias[n].
// grid.z batches via aBS/bBS/cBS element strides.  Requires K % 64 == 0.
// -------------------------------------------------------------------------
template<bool OUT_BF16, bool TRANS, bool BIAS>
__global__ __launch_bounds__(256)
void wmma_gemm(const unsigned short* __restrict__ A,
               const unsigned short* __restrict__ Bm,
               const float* __restrict__ bias,
               void* __restrict__ Cout,
               int M, int N, int K,
               long aBS, long bBS, long cBS,
               float scale, int trans_rows, long trans_bstride)
{
    const int lane = threadIdx.x & 31;
    const int wave = threadIdx.x >> 5;
    const int wm = wave >> 2;          // 0..1
    const int wn = wave & 3;           // 0..3
    const int z  = blockIdx.z;

    gc_us* Ag = (gc_us*)(A  + (long)z * aBS);
    gc_us* Bg = (gc_us*)(Bm + (long)z * bBS);
    gc_f32* biasg = (gc_f32*)bias;

    const int m_w = blockIdx.y * 128 + wm * 64;
    const int n_w = blockIdx.x * 128 + wn * 32;

    const int rowLane = lane & 15;
    const int kPhase  = (lane & 16) ? 8 : 0;

    // Hoisted per-lane fragment base pointers (advanced by 64 elems per pair).
    gc_us* pa[4];
    gc_us* pb[2];
#pragma unroll
    for (int ti = 0; ti < 4; ++ti)
        pa[ti] = Ag + (long)(m_w + ti * 16 + rowLane) * K + kPhase;
#pragma unroll
    for (int tj = 0; tj < 2; ++tj)
        pb[tj] = Bg + (long)(n_w + tj * 16 + rowLane) * K + kPhase;

    const v8f zero8 = {0.f, 0.f, 0.f, 0.f, 0.f, 0.f, 0.f, 0.f};
    v8f acc[4][2];
#pragma unroll
    for (int ti = 0; ti < 4; ++ti)
#pragma unroll
        for (int tj = 0; tj < 2; ++tj) acc[ti][tj] = zero8;

    auto mm = [&](const v16bf (&aa)[4], const v16bf (&bb)[2]) {
#pragma unroll
        for (int ti = 0; ti < 4; ++ti)
#pragma unroll
            for (int tj = 0; tj < 2; ++tj)
                acc[ti][tj] = __builtin_amdgcn_wmma_f32_16x16x32_bf16(
                    false, aa[ti], false, bb[tj], (short)0, acc[ti][tj],
                    false, false);
    };

    v16bf a0[4], b0[2], a1[4], b1[2];

    // prologue: K-step 0
#pragma unroll
    for (int ti = 0; ti < 4; ++ti) a0[ti] = load_frag_g(pa[ti], 0);
#pragma unroll
    for (int tj = 0; tj < 2; ++tj) b0[tj] = load_frag_g(pb[tj], 0);

    const int steps = K >> 5;          // K/32, even for all our shapes
    for (int t = 0; t < steps - 2; t += 2) {
        // load step t+1 (immediate offset +64B) while computing step t
#pragma unroll
        for (int ti = 0; ti < 4; ++ti) a1[ti] = load_frag_g(pa[ti], 32);
#pragma unroll
        for (int tj = 0; tj < 2; ++tj) b1[tj] = load_frag_g(pb[tj], 32);
        mm(a0, b0);
        // advance to step t+2 and load it while computing step t+1
#pragma unroll
        for (int ti = 0; ti < 4; ++ti) pa[ti] += 64;
#pragma unroll
        for (int tj = 0; tj < 2; ++tj) pb[tj] += 64;
#pragma unroll
        for (int ti = 0; ti < 4; ++ti) a0[ti] = load_frag_g(pa[ti], 0);
#pragma unroll
        for (int tj = 0; tj < 2; ++tj) b0[tj] = load_frag_g(pb[tj], 0);
        mm(a1, b1);
    }
    // epilogue: steps-2 in buf0, load steps-1, compute both
#pragma unroll
    for (int ti = 0; ti < 4; ++ti) a1[ti] = load_frag_g(pa[ti], 32);
#pragma unroll
    for (int tj = 0; tj < 2; ++tj) b1[tj] = load_frag_g(pb[tj], 32);
    mm(a0, b0);
    mm(a1, b1);

    // C/D layout: vgpr i, lane<16 -> (m = i,     n = lane)
    //                     lane>=16 -> (m = i + 8, n = lane-16)
#pragma unroll
    for (int ti = 0; ti < 4; ++ti) {
#pragma unroll
        for (int tj = 0; tj < 2; ++tj) {
            const int n     = n_w + tj * 16 + rowLane;
            const int mbase = m_w + ti * 16 + ((lane & 16) ? 8 : 0);
            const float bi  = BIAS ? biasg[n] : 0.0f;
            if (TRANS) {
                // 8 contiguous m per lane -> one 16B bf16 store into C^T
                v4u w;
#pragma unroll
                for (int j = 0; j < 4; ++j) {
                    unsigned short lo = f2bf(acc[ti][tj][2 * j]     * scale + bi);
                    unsigned short hi = f2bf(acc[ti][tj][2 * j + 1] * scale + bi);
                    w[j] = (unsigned int)lo | ((unsigned int)hi << 16);
                }
                const int gb = mbase / trans_rows;
                const int ml = mbase - gb * trans_rows;
                g_us* dst = (g_us*)Cout +
                    (long)gb * trans_bstride + (long)n * trans_rows + ml;
                *(g_v4u*)dst = w;
            } else if (OUT_BF16) {
                g_us* dst = (g_us*)Cout + (long)z * cBS;
#pragma unroll
                for (int i = 0; i < 8; ++i)
                    dst[(long)(mbase + i) * N + n] =
                        f2bf(acc[ti][tj][i] * scale + bi);
            } else {
                g_f32* dst = (g_f32*)Cout + (long)z * cBS;
#pragma unroll
                for (int i = 0; i < 8; ++i)
                    dst[(long)(mbase + i) * N + n] =
                        acc[ti][tj][i] * scale + bi;
            }
        }
    }
}

// -------------------------------------------------------------------------
// fp32 -> bf16 (RNE), 8 elements per thread, vectorized 16B in / 16B out
// -------------------------------------------------------------------------
__global__ __launch_bounds__(256)
void f32_to_bf16_kernel(const float* __restrict__ in,
                        unsigned short* __restrict__ out, long n)
{
    long i = ((long)blockIdx.x * blockDim.x + threadIdx.x) * 8;
    if (i >= n) return;
    gc_v4f* p = (gc_v4f*)(in + i);
    v4f a = p[0], b = p[1];
    v4u w;
    w[0] = (unsigned int)f2bf(a[0]) | ((unsigned int)f2bf(a[1]) << 16);
    w[1] = (unsigned int)f2bf(a[2]) | ((unsigned int)f2bf(a[3]) << 16);
    w[2] = (unsigned int)f2bf(b[0]) | ((unsigned int)f2bf(b[1]) << 16);
    w[3] = (unsigned int)f2bf(b[2]) | ((unsigned int)f2bf(b[3]) << 16);
    *(g_v4u*)((g_us*)out + i) = w;
}

// -------------------------------------------------------------------------
// Row softmax over bf16 rows of length 2048, in place.
// One wave32 per row (8 rows per 256-thread block); 64 elems per lane.
// -------------------------------------------------------------------------
__global__ __launch_bounds__(256)
void softmax_rows(unsigned short* __restrict__ P, int rows, int cols)
{
    const int lane = threadIdx.x & 31;
    const int wave = threadIdx.x >> 5;
    const int row  = blockIdx.x * 8 + wave;
    if (row >= rows) return;

    g_us* rp = (g_us*)P + (long)row * cols;
    gc_v4u* vp = (gc_v4u*)rp;

    float x[64];
#pragma unroll
    for (int c = 0; c < 8; ++c) {
        v4u q = vp[lane + 32 * c];
#pragma unroll
        for (int j = 0; j < 4; ++j) {
            x[c * 8 + 2 * j]     = bf2f((unsigned short)(q[j] & 0xFFFFu));
            x[c * 8 + 2 * j + 1] = bf2f((unsigned short)(q[j] >> 16));
        }
    }

    float m = -3.0e38f;
#pragma unroll
    for (int i = 0; i < 64; ++i) m = fmaxf(m, x[i]);
#pragma unroll
    for (int off = 16; off > 0; off >>= 1)
        m = fmaxf(m, __shfl_xor(m, off, 32));

    float s = 0.f;
#pragma unroll
    for (int i = 0; i < 64; ++i) { x[i] = __expf(x[i] - m); s += x[i]; }
#pragma unroll
    for (int off = 16; off > 0; off >>= 1)
        s += __shfl_xor(s, off, 32);

    const float inv = 1.0f / s;
    g_v4u* ovp = (g_v4u*)rp;
#pragma unroll
    for (int c = 0; c < 8; ++c) {
        v4u w;
#pragma unroll
        for (int j = 0; j < 4; ++j) {
            unsigned short lo = f2bf(x[c * 8 + 2 * j]     * inv);
            unsigned short hi = f2bf(x[c * 8 + 2 * j + 1] * inv);
            w[j] = (unsigned int)lo | ((unsigned int)hi << 16);
        }
        ovp[lane + 32 * c] = w;
    }
}

// -------------------------------------------------------------------------
extern "C" void kernel_launch(void* const* d_in, const int* in_sizes, int n_in,
                              void* d_out, int out_size, void* d_ws, size_t ws_size,
                              hipStream_t stream)
{
    (void)in_sizes; (void)n_in; (void)out_size; (void)ws_size;

    const float* x  = (const float*)d_in[0];
    const float* Wq = (const float*)d_in[1];
    const float* bq = (const float*)d_in[2];
    const float* Wk = (const float*)d_in[3];
    const float* bk = (const float*)d_in[4];
    const float* Wv = (const float*)d_in[5];
    const float* bv = (const float*)d_in[6];
    const float* Wo = (const float*)d_in[7];
    const float* bo = (const float*)d_in[8];
    float* out = (float*)d_out;

    const long MS  = (long)B_ * S_;          // 8192 total rows
    const long XN  = MS * D_;                // 8,388,608 elems
    const long WN  = (long)D_ * D_;          // 1,048,576 elems
    const long PN  = (long)B_ * S_ * S_;     // 16,777,216 elems

    // workspace layout (bf16 buffers, all 16B aligned)
    char* w = (char*)d_ws;
    unsigned short* xb   = (unsigned short*)w; w += XN * 2;
    unsigned short* Wqb  = (unsigned short*)w; w += WN * 2;
    unsigned short* Wkb  = (unsigned short*)w; w += WN * 2;
    unsigned short* Wvb  = (unsigned short*)w; w += WN * 2;
    unsigned short* Wob  = (unsigned short*)w; w += WN * 2;
    unsigned short* Qb   = (unsigned short*)w; w += XN * 2;
    unsigned short* Kb   = (unsigned short*)w; w += XN * 2;
    unsigned short* Vtb  = (unsigned short*)w; w += XN * 2;   // per-batch [D][S]
    unsigned short* Pb   = (unsigned short*)w; w += PN * 2;
    unsigned short* ctxb = (unsigned short*)w; w += XN * 2;

    const dim3 blk(256);

    // 1) fp32 -> bf16 conversions
    f32_to_bf16_kernel<<<dim3((unsigned)(XN / 8 / 256)), blk, 0, stream>>>(x,  xb,  XN);
    f32_to_bf16_kernel<<<dim3((unsigned)(WN / 8 / 256)), blk, 0, stream>>>(Wq, Wqb, WN);
    f32_to_bf16_kernel<<<dim3((unsigned)(WN / 8 / 256)), blk, 0, stream>>>(Wk, Wkb, WN);
    f32_to_bf16_kernel<<<dim3((unsigned)(WN / 8 / 256)), blk, 0, stream>>>(Wv, Wvb, WN);
    f32_to_bf16_kernel<<<dim3((unsigned)(WN / 8 / 256)), blk, 0, stream>>>(Wo, Wob, WN);

    // 2) QKV projections: [8192 x 1024] = x[8192x1024] . W^T[1024x1024] + b
    {
        dim3 grid(D_ / 128, (unsigned)(MS / 128), 1);
        wmma_gemm<true, false, true><<<grid, blk, 0, stream>>>(
            xb, Wqb, bq, Qb, (int)MS, D_, D_, 0, 0, 0, 1.0f, 1, 0);
        wmma_gemm<true, false, true><<<grid, blk, 0, stream>>>(
            xb, Wkb, bk, Kb, (int)MS, D_, D_, 0, 0, 0, 1.0f, 1, 0);
        // V projection stored transposed per batch: Vtb[b][d][s]
        wmma_gemm<true, true, true><<<grid, blk, 0, stream>>>(
            xb, Wvb, bv, Vtb, (int)MS, D_, D_, 0, 0, 0, 1.0f,
            S_, (long)D_ * S_);
    }

    // 3) scores = (Q . K^T) / sqrt(D), per batch, bf16 out
    {
        dim3 grid(S_ / 128, S_ / 128, B_);
        wmma_gemm<true, false, false><<<grid, blk, 0, stream>>>(
            Qb, Kb, nullptr, Pb, S_, S_, D_,
            (long)S_ * D_, (long)S_ * D_, (long)S_ * S_,
            0.03125f /* 1/sqrt(1024) */, 1, 0);
    }

    // 4) softmax rows in place
    softmax_rows<<<dim3((unsigned)(MS / 8)), blk, 0, stream>>>(Pb, (int)MS, S_);

    // 5) ctx = P . V   (B-operand rows are rows of V^T), bf16 out
    {
        dim3 grid(D_ / 128, S_ / 128, B_);
        wmma_gemm<true, false, false><<<grid, blk, 0, stream>>>(
            Pb, Vtb, nullptr, ctxb, S_, D_, S_,
            (long)S_ * S_, (long)D_ * S_, (long)S_ * D_,
            1.0f, 1, 0);
    }

    // 6) out = ctx . Wo^T + bo, fp32 out
    {
        dim3 grid(D_ / 128, (unsigned)(MS / 128), 1);
        wmma_gemm<false, false, true><<<grid, blk, 0, stream>>>(
            ctxb, Wob, bo, out, (int)MS, D_, D_, 0, 0, 0, 1.0f, 1, 0);
    }
}
